// ContrastPoolLayer_67070209295064
// MI455X (gfx1250) — matrix operand order repaired and checked by reading, not compiled, not run on hardware
//
#include <hip/hip_runtime.h>
#include <hip/hip_fp16.h>

// Problem constants (match reference)
#define NB      32        // graphs
#define NPER    512       // nodes per graph
#define NN      16384     // total nodes
#define NE      262144    // edges
#define DIN     128
#define DOUT    128
#define NA      2048      // total clusters
#define KC      64        // clusters per graph

typedef __attribute__((ext_vector_type(16))) _Float16 v16h;
typedef __attribute__((ext_vector_type(8)))  _Float16 v8h;
typedef __attribute__((ext_vector_type(4)))  _Float16 v4h;
typedef __attribute__((ext_vector_type(8)))  float    v8f;
typedef __attribute__((ext_vector_type(4)))  int      v4i;

// ---------------- async global->LDS staging (CDNA5), with sync fallback -------------
#if defined(__has_builtin)
#if __has_builtin(__builtin_amdgcn_global_load_async_to_lds_b128)
#define HAS_ASYNC_LDS 1
#endif
#endif
#ifndef HAS_ASYNC_LDS
#define HAS_ASYNC_LDS 0
#endif

#if HAS_ASYNC_LDS
typedef __attribute__((address_space(1))) v4i gv4i_t;   // global int4
typedef __attribute__((address_space(3))) v4i sv4i_t;   // LDS int4
#endif

// copy 8 halves (16B) global -> LDS
__device__ __forceinline__ void cp16(_Float16* l, const _Float16* g) {
#if HAS_ASYNC_LDS
    __builtin_amdgcn_global_load_async_to_lds_b128(
        (gv4i_t*)(void*)g, (sv4i_t*)(void*)l, 0, 0);
#else
    *(v8h*)l = *(const v8h*)g;
#endif
}

__device__ __forceinline__ void stage_wait() {
#if HAS_ASYNC_LDS
#if __has_builtin(__builtin_amdgcn_s_wait_asynccnt)
    __builtin_amdgcn_s_wait_asynccnt(0);
#else
    asm volatile("s_wait_asynccnt 0" ::: "memory");
#endif
#endif
    __syncthreads();
}

// ---------------- WMMA fragment loaders (wave32, 16x16x32 f16) ----------------
// A tile in LDS: row-major [M][32] halves. Lane L: m = mBase + (L&15), group
// g = L>>4; elements 0..7 -> K = g*8+0..7, elements 8..15 -> K = 16+g*8+0..7.
__device__ inline v16h ldA_frag(const _Float16* sA, int mBase, int lane) {
    int m = mBase + (lane & 15);
    int g = lane >> 4;
    const _Float16* p = sA + m * 32 + g * 8;
    v8h lo = *(const v8h*)(p);
    v8h hi = *(const v8h*)(p + 16);
    return __builtin_shufflevector(lo, hi, 0,1,2,3,4,5,6,7,8,9,10,11,12,13,14,15);
}
// B tile staged K-contiguous per column: [N][32] halves.
// Lane L: n = nBase + (L&15), elements 0..7 -> K = (L>>4)*8+e, 8..15 -> K = 16+(L>>4)*8+(e-8).
__device__ inline v16h ldB_frag(const _Float16* sBt, int nBase, int lane) {
    int n = nBase + (lane & 15);
    int g = lane >> 4;
    const _Float16* p = sBt + n * 32 + g * 8;
    v8h lo = *(const v8h*)(p);
    v8h hi = *(const v8h*)(p + 16);
    return __builtin_shufflevector(lo, hi, 0,1,2,3,4,5,6,7,8,9,10,11,12,13,14,15);
}
#define WMMA_F16(a, b, c) \
    __builtin_amdgcn_wmma_f32_16x16x32_f16(false, (a), false, (b), (short)0, (c), false, false)

// ---------------- utility kernels ----------------
__global__ void fill_f32(float* p, float v, int n) {
    for (int i = blockIdx.x * blockDim.x + threadIdx.x; i < n; i += gridDim.x * blockDim.x)
        p[i] = v;
}

// W [256][Nn] f32 row-major  ->  WT [Nn][256] f16 (K-contiguous rows)
__global__ __launch_bounds__(256) void transpose_w(const float* __restrict__ W,
                                                   _Float16* __restrict__ WT, int Nn) {
    int t = blockIdx.x * 256 + threadIdx.x;
    if (t >= Nn * 32) return;
    int k8 = (t / Nn) * 8;
    int n  = t % Nn;
    v8h o;
#pragma unroll
    for (int j = 0; j < 8; ++j) o[j] = (_Float16)W[(size_t)(k8 + j) * Nn + n];
    *(v8h*)&WT[(size_t)n * 256 + k8] = o;
}

// a_s [NN][64] f32  ->  a_sT [64][NN] f16
__global__ __launch_bounds__(256) void transpose_as(const float* __restrict__ a_s,
                                                    _Float16* __restrict__ a_sT) {
    int t = blockIdx.x * 256 + threadIdx.x;
    if (t >= NN * KC) return;
    int node = t >> 6, c = t & 63;
    a_sT[(size_t)c * NN + node] = (_Float16)a_s[(size_t)node * KC + c];
}

// ---------------- edge aggregation: msg[dst] += h[src]; deg[dst] += 1 ----------------
__global__ __launch_bounds__(256) void edge_aggregate(const float* __restrict__ h,
                                                      const int* __restrict__ esrc,
                                                      const int* __restrict__ edst,
                                                      float* __restrict__ msg,
                                                      float* __restrict__ deg) {
    int t = blockIdx.x * 256 + threadIdx.x;
    int e = t >> 5, q = t & 31;
    if (e >= NE) return;
    int s = esrc[e], d = edst[e];
    float4 v = *(const float4*)&h[(size_t)s * DIN + q * 4];
    float* mp = &msg[(size_t)d * DIN + q * 4];
    unsafeAtomicAdd(mp + 0, v.x);
    unsafeAtomicAdd(mp + 1, v.y);
    unsafeAtomicAdd(mp + 2, v.z);
    unsafeAtomicAdd(mp + 3, v.w);
    if (q == 0) unsafeAtomicAdd(&deg[d], 1.0f);
}

// ---------------- X = [h | msg/max(deg,1)] as f16, row-major [NN][256] ----------------
__global__ __launch_bounds__(256) void build_x16(const float* __restrict__ h,
                                                 const float* __restrict__ msg,
                                                 const float* __restrict__ deg,
                                                 _Float16* __restrict__ X) {
    int t = blockIdx.x * 256 + threadIdx.x;
    int i = t >> 6;
    int c4 = (t & 63) * 4;
    if (i >= NN) return;
    v4h o;
    if (c4 < DIN) {
        float4 v = *(const float4*)&h[(size_t)i * DIN + c4];
        o[0] = (_Float16)v.x; o[1] = (_Float16)v.y; o[2] = (_Float16)v.z; o[3] = (_Float16)v.w;
    } else {
        float invd = 1.0f / fmaxf(deg[i], 1.0f);
        float4 v = *(const float4*)&msg[(size_t)i * DIN + (c4 - DIN)];
        o[0] = (_Float16)(v.x * invd); o[1] = (_Float16)(v.y * invd);
        o[2] = (_Float16)(v.z * invd); o[3] = (_Float16)(v.w * invd);
    }
    *(v4h*)&X[(size_t)i * 256 + c4] = o;
}

// ---------------- GEMM feat: [16384,128] = X @ Wf, fused bias+L2norm+relu --------------
// Emits featT f16 [128][16384] (K-contiguous for the h_new GEMM).
__global__ __launch_bounds__(256) void gemm_feat(const _Float16* __restrict__ X,
                                                 const _Float16* __restrict__ WT,
                                                 const float* __restrict__ bias,
                                                 _Float16* __restrict__ featT) {
    __shared__ _Float16 sA[64 * 32];
    __shared__ _Float16 sBt[128 * 32];
    __shared__ float    sC[64 * 128];
    __shared__ float    sScale[64];
    const int tid = threadIdx.x, lane = tid & 31, wave = tid >> 5;
    const int m0 = blockIdx.x * 64;
    const int msub = (wave & 3) * 16;
    const int nsub = (wave >> 2) * 64;
    v8f acc[4] = {};
    for (int k0 = 0; k0 < 256; k0 += 32) {
        { // A: 64 rows x 32 halves = 256 x 16B chunks
            int r = tid >> 2, c = (tid & 3) * 8;
            cp16(&sA[r * 32 + c], &X[(size_t)(m0 + r) * 256 + k0 + c]);
            if (k0 + 32 < 256)
                __builtin_prefetch(&X[(size_t)(m0 + r) * 256 + k0 + 32 + c], 0, 1);
        }
        // B: 128 cols x 32 halves = 512 x 16B chunks (WT is K-contiguous)
        for (int idx = tid; idx < 128 * 4; idx += 256) {
            int n = idx >> 2, c = (idx & 3) * 8;
            cp16(&sBt[n * 32 + c], &WT[(size_t)n * 256 + k0 + c]);
        }
        stage_wait();
        v16h a = ldA_frag(sA, msub, lane);
#pragma unroll
        for (int j = 0; j < 4; ++j) {
            v16h b = ldB_frag(sBt, nsub + j * 16, lane);
            acc[j] = WMMA_F16(a, b, acc[j]);
        }
        __syncthreads();
    }
#pragma unroll
    for (int j = 0; j < 4; ++j) {
        int n = nsub + j * 16 + (lane & 15);
#pragma unroll
        for (int r = 0; r < 8; ++r) {
            int m = msub + (lane >> 4) * 8 + r;
            sC[m * 128 + n] = acc[j][r] + bias[n];
        }
    }
    __syncthreads();
    { // row L2 norms (4 threads/row x 32 cols)
        int row = tid >> 2, part = tid & 3;
        float ssq = 0.f;
#pragma unroll
        for (int c = 0; c < 32; ++c) {
            float v = sC[row * 128 + part * 32 + c];
            ssq += v * v;
        }
        ssq += __shfl_xor(ssq, 1);
        ssq += __shfl_xor(ssq, 2);
        if (part == 0) sScale[row] = 1.0f / fmaxf(sqrtf(ssq), 1e-12f);
    }
    __syncthreads();
    // transposed f16 store (col-major walk -> contiguous halves along rows)
    for (int idx = tid; idx < 64 * 128; idx += 256) {
        int col = idx >> 6, row = idx & 63;
        float v = fmaxf(sC[row * 128 + col] * sScale[row], 0.f);
        featT[(size_t)col * NN + m0 + row] = (_Float16)v;
    }
}

// ---------------- GEMM pool: full [16384,2048] for norms; emit in-block 64 cols -------
__global__ __launch_bounds__(256) void gemm_pool(const _Float16* __restrict__ X,
                                                 const _Float16* __restrict__ WT,
                                                 const float* __restrict__ bias,
                                                 float* __restrict__ poolc,
                                                 float* __restrict__ rowssq) {
    __shared__ _Float16 sA[64 * 32];
    __shared__ _Float16 sBt[64 * 32];
    __shared__ float    sC[64 * 64];
    const int tid = threadIdx.x, lane = tid & 31, wave = tid >> 5;
    const int m0 = blockIdx.x * 64;
    const int n0 = blockIdx.y * 64;
    const int msub = (wave & 3) * 16;
    const int nsub = (wave >> 2) * 32;
    v8f acc[2] = {};
    for (int k0 = 0; k0 < 256; k0 += 32) {
        {
            int r = tid >> 2, c = (tid & 3) * 8;
            cp16(&sA[r * 32 + c], &X[(size_t)(m0 + r) * 256 + k0 + c]);
            if (k0 + 32 < 256)
                __builtin_prefetch(&X[(size_t)(m0 + r) * 256 + k0 + 32 + c], 0, 1);
        }
        { // B: 64 cols x 4 chunks = exactly 256 threads
            int n = tid >> 2, c = (tid & 3) * 8;
            cp16(&sBt[n * 32 + c], &WT[(size_t)(n0 + n) * 256 + k0 + c]);
        }
        stage_wait();
        v16h a = ldA_frag(sA, msub, lane);
#pragma unroll
        for (int j = 0; j < 2; ++j) {
            v16h b = ldB_frag(sBt, nsub + j * 16, lane);
            acc[j] = WMMA_F16(a, b, acc[j]);
        }
        __syncthreads();
    }
#pragma unroll
    for (int j = 0; j < 2; ++j) {
        int n = nsub + j * 16 + (lane & 15);
#pragma unroll
        for (int r = 0; r < 8; ++r) {
            int m = msub + (lane >> 4) * 8 + r;
            sC[m * 64 + n] = acc[j][r] + bias[n0 + n];
        }
    }
    __syncthreads();
    int row = tid >> 2, part = tid & 3;
    float ssq = 0.f;
#pragma unroll
    for (int c = 0; c < 16; ++c) {
        float v = sC[row * 64 + part * 16 + c];
        ssq += v * v;
    }
    ssq += __shfl_xor(ssq, 1);
    ssq += __shfl_xor(ssq, 2);
    if (part == 0) unsafeAtomicAdd(&rowssq[m0 + row], ssq);
    if ((int)blockIdx.y == (m0 / NPER)) {   // only the in-graph tile survives the mask
#pragma unroll
        for (int c = 0; c < 16; ++c)
            poolc[(size_t)(m0 + row) * KC + part * 16 + c] = sC[row * 64 + part * 16 + c];
    }
}

// ---------------- masked softmax over 64 in-block clusters (1 wave32 / node) ---------
// Writes s row-major (for the edge scatter) and transposed (for the GEMMs).
__global__ __launch_bounds__(256) void softmax_s(const float* __restrict__ poolc,
                                                 const float* __restrict__ rowssq,
                                                 _Float16* __restrict__ s16,
                                                 _Float16* __restrict__ s16T) {
    int i = blockIdx.x * 8 + (threadIdx.x >> 5);
    int lane = threadIdx.x & 31;
    if (i >= NN) return;
    float inv = 1.0f / fmaxf(sqrtf(rowssq[i]), 1e-12f);
    float a0 = fmaxf(poolc[(size_t)i * KC + lane] * inv, 0.f);
    float a1 = fmaxf(poolc[(size_t)i * KC + 32 + lane] * inv, 0.f);
    float m = fmaxf(a0, a1);
#pragma unroll
    for (int x = 16; x; x >>= 1) m = fmaxf(m, __shfl_xor(m, x));
    float e0 = __expf(a0 - m), e1 = __expf(a1 - m);
    float s = e0 + e1;
#pragma unroll
    for (int x = 16; x; x >>= 1) s += __shfl_xor(s, x);
    float r = 1.0f / s;
    _Float16 h0 = (_Float16)(e0 * r), h1 = (_Float16)(e1 * r);
    s16[(size_t)i * KC + lane]       = h0;
    s16[(size_t)i * KC + 32 + lane]  = h1;
    s16T[(size_t)lane * NN + i]        = h0;
    s16T[(size_t)(lane + 32) * NN + i] = h1;
}

// ---------------- a_s[dst] += s[src] (compact 64-wide), 16 lanes/edge ----------------
__global__ __launch_bounds__(256) void edge_as(const _Float16* __restrict__ s16,
                                               const int* __restrict__ esrc,
                                               const int* __restrict__ edst,
                                               float* __restrict__ a_s) {
    int t = blockIdx.x * 256 + threadIdx.x;
    int e = t >> 4, q = t & 15;
    if (e >= NE) return;
    int s = esrc[e], d = edst[e];
    v4h v = *(const v4h*)&s16[(size_t)s * KC + q * 4];
    float* p = &a_s[(size_t)d * KC + q * 4];
    unsafeAtomicAdd(p + 0, (float)v[0]);
    unsafeAtomicAdd(p + 1, (float)v[1]);
    unsafeAtomicAdd(p + 2, (float)v[2]);
    unsafeAtomicAdd(p + 3, (float)v[3]);
}

// ---------------- h_new block g: [64,128] = s_g^T[64,512] @ feat_g[512,128] ----------
__global__ __launch_bounds__(256) void gemm_hnew(const _Float16* __restrict__ s16T,
                                                 const _Float16* __restrict__ featT,
                                                 float* __restrict__ out) {
    __shared__ _Float16 sA[64 * 32];
    __shared__ _Float16 sBt[128 * 32];
    const int g = blockIdx.x;
    const size_t node0 = (size_t)g * NPER;
    const int tid = threadIdx.x, lane = tid & 31, wave = tid >> 5;
    const int msub = (wave & 3) * 16;
    const int nsub = (wave >> 2) * 64;
    v8f acc[4] = {};
    for (int k0 = 0; k0 < NPER; k0 += 32) {
        { // A (s^T): 64 clusters x 32 nodes, K-contiguous in s16T
            int c = tid >> 2, cc = (tid & 3) * 8;
            cp16(&sA[c * 32 + cc], &s16T[(size_t)c * NN + node0 + k0 + cc]);
        }
        for (int idx = tid; idx < 128 * 4; idx += 256) {
            int n = idx >> 2, cc = (idx & 3) * 8;
            cp16(&sBt[n * 32 + cc], &featT[(size_t)n * NN + node0 + k0 + cc]);
        }
        stage_wait();
        v16h a = ldA_frag(sA, msub, lane);
#pragma unroll
        for (int j = 0; j < 4; ++j) {
            v16h b = ldB_frag(sBt, nsub + j * 16, lane);
            acc[j] = WMMA_F16(a, b, acc[j]);
        }
        __syncthreads();
    }
#pragma unroll
    for (int j = 0; j < 4; ++j) {
        int n = nsub + j * 16 + (lane & 15);
#pragma unroll
        for (int r = 0; r < 8; ++r) {
            int m = msub + (lane >> 4) * 8 + r;
            out[(size_t)(g * KC + m) * DOUT + n] = acc[j][r];
        }
    }
}

// ---------------- adj block g: [64,64] = s_g^T @ a_s_g -> diagonal block -------------
__global__ __launch_bounds__(256) void gemm_adj(const _Float16* __restrict__ s16T,
                                                const _Float16* __restrict__ a_sT,
                                                float* __restrict__ out) {
    __shared__ _Float16 sA[64 * 32];
    __shared__ _Float16 sBt[64 * 32];
    const int g = blockIdx.x;
    const size_t node0 = (size_t)g * NPER;
    const int tid = threadIdx.x, lane = tid & 31, wave = tid >> 5;
    const int msub = (wave & 3) * 16;
    const int nsub = (wave >> 2) * 32;
    v8f acc[2] = {};
    for (int k0 = 0; k0 < NPER; k0 += 32) {
        {
            int c = tid >> 2, cc = (tid & 3) * 8;
            cp16(&sA[c * 32 + cc], &s16T[(size_t)c * NN + node0 + k0 + cc]);
        }
        {
            int n = tid >> 2, cc = (tid & 3) * 8;
            cp16(&sBt[n * 32 + cc], &a_sT[(size_t)n * NN + node0 + k0 + cc]);
        }
        stage_wait();
        v16h a = ldA_frag(sA, msub, lane);
#pragma unroll
        for (int j = 0; j < 2; ++j) {
            v16h b = ldB_frag(sBt, nsub + j * 16, lane);
            acc[j] = WMMA_F16(a, b, acc[j]);
        }
        __syncthreads();
    }
#pragma unroll
    for (int j = 0; j < 2; ++j) {
        int n = nsub + j * 16 + (lane & 15);
#pragma unroll
        for (int r = 0; r < 8; ++r) {
            int m = msub + (lane >> 4) * 8 + r;
            out[(size_t)(g * KC + m) * NA + g * KC + n] = acc[j][r];
        }
    }
}

// ---------------- launch ----------------
extern "C" void kernel_launch(void* const* d_in, const int* in_sizes, int n_in,
                              void* d_out, int out_size, void* d_ws, size_t ws_size,
                              hipStream_t stream) {
    (void)in_sizes; (void)n_in; (void)out_size; (void)ws_size;
    const float* h      = (const float*)d_in[0];
    const float* W_feat = (const float*)d_in[1];
    const float* b_feat = (const float*)d_in[2];
    const float* W_pool = (const float*)d_in[3];
    const float* b_pool = (const float*)d_in[4];
    const int*   esrc   = (const int*)d_in[5];
    const int*   edst   = (const int*)d_in[6];
    float* out = (float*)d_out;

    char* ws = (char*)d_ws;
    float*    msg    = (float*)ws;    ws += (size_t)NN * DIN * 4;    // 8 MB
    float*    deg    = (float*)ws;    ws += (size_t)NN * 4;          // 64 KB
    float*    rowssq = (float*)ws;    ws += (size_t)NN * 4;          // 64 KB
    float*    a_s    = (float*)ws;    ws += (size_t)NN * KC * 4;     // 4 MB
    _Float16* X16    = (_Float16*)ws; ws += (size_t)NN * 256 * 2;    // 8 MB
    _Float16* WfT    = (_Float16*)ws; ws += (size_t)128 * 256 * 2;   // 64 KB
    _Float16* WpT    = (_Float16*)ws; ws += (size_t)NA * 256 * 2;    // 1 MB
    _Float16* featT  = (_Float16*)ws; ws += (size_t)DOUT * NN * 2;   // 4 MB
    float*    poolc  = (float*)ws;    ws += (size_t)NN * KC * 4;     // 4 MB
    _Float16* s16    = (_Float16*)ws; ws += (size_t)NN * KC * 2;     // 2 MB
    _Float16* s16T   = (_Float16*)ws; ws += (size_t)KC * NN * 2;     // 2 MB
    _Float16* a_sT   = (_Float16*)ws; ws += (size_t)KC * NN * 2;     // 2 MB

    const int TB = 256;
    fill_f32<<<2048, TB, 0, stream>>>(msg, 0.f, NN * DIN);
    fill_f32<<<64,   TB, 0, stream>>>(deg, 0.f, NN);
    fill_f32<<<64,   TB, 0, stream>>>(rowssq, 0.f, NN);
    fill_f32<<<2048, TB, 0, stream>>>(a_s, 0.f, NN * KC);
    fill_f32<<<4096, TB, 0, stream>>>(out, 0.f, NA * NA);  // zero off-diagonal adj

    edge_aggregate<<<(NE * 32) / TB, TB, 0, stream>>>(h, esrc, edst, msg, deg);
    build_x16<<<(NN * 64) / TB, TB, 0, stream>>>(h, msg, deg, X16);
    transpose_w<<<(128 * 32) / TB, TB, 0, stream>>>(W_feat, WfT, 128);
    transpose_w<<<(NA * 32) / TB,  TB, 0, stream>>>(W_pool, WpT, NA);

    gemm_feat<<<NN / 64, TB, 0, stream>>>(X16, WfT, b_feat, featT);
    dim3 gp(NN / 64, NA / 64);
    gemm_pool<<<gp, TB, 0, stream>>>(X16, WpT, b_pool, poolc, rowssq);
    softmax_s<<<NN / 8, TB, 0, stream>>>(poolc, rowssq, s16, s16T);
    edge_as<<<(NE * 16) / TB, TB, 0, stream>>>(s16, esrc, edst, a_s);
    transpose_as<<<(NN * KC) / TB, TB, 0, stream>>>(a_s, a_sT);

    gemm_hnew<<<NB, TB, 0, stream>>>(s16T, featT, out + (size_t)NA * NA);
    gemm_adj<<<NB, TB, 0, stream>>>(s16T, a_sT, out);
}
